// Encoder_27590869910160
// MI455X (gfx1250) — matrix operand order repaired and checked by reading, not compiled, not run on hardware
//
#include <hip/hip_runtime.h>
#include <hip/hip_bf16.h>
#include <math.h>

typedef __attribute__((ext_vector_type(16))) _Float16 v16h;
typedef __attribute__((ext_vector_type(8)))  _Float16 v8h;
typedef __attribute__((ext_vector_type(4)))  _Float16 v4h;
typedef __attribute__((ext_vector_type(8)))  float    v8f;

#define BN_EPS 1e-3f

// ---------------- small utility kernels ----------------

// Convert W[K,P] f32 (row-major) -> Wt[P,K] f16 (i.e. column-major of W).
// Makes WMMA B-fragment loads contiguous (two b128 loads per fragment).
__global__ void k_w_to_f16_t(const float* __restrict__ in, _Float16* __restrict__ out,
                             int K, int P) {
  int i = blockIdx.x * blockDim.x + threadIdx.x;
  if (i >= K * P) return;
  int k = i / P, p = i - k * P;
  out[(size_t)p * K + k] = (_Float16)in[i];
}

// BN1 over x[N,128], emit f16 for WMMA GEMM input.
__global__ void k_bn1(const float* __restrict__ x,
                      const float* __restrict__ g, const float* __restrict__ b,
                      const float* __restrict__ m, const float* __restrict__ v,
                      _Float16* __restrict__ out, int total) {
  int i = blockIdx.x * blockDim.x + threadIdx.x;
  if (i >= total) return;
  int c = i & 127;
  float val = g[c] * (x[i] - m[c]) * rsqrtf(v[c] + BN_EPS) + b[c];
  out[i] = (_Float16)val;
}

// edge_dst is sorted: row_start[n] = lower_bound(dst, n). row_start[N] = E.
__global__ void k_rowstart(const int* __restrict__ dst, int* __restrict__ row_start,
                           int N, int E) {
  int n = blockIdx.x * blockDim.x + threadIdx.x;
  if (n > N) return;
  int lo = 0, hi = E;
  while (lo < hi) { int mid = (lo + hi) >> 1; if (dst[mid] < n) lo = mid + 1; else hi = mid; }
  row_start[n] = lo;
}

// ---------------- WMMA GEMM: C[N,P] = A_f16[N,K] @ W[K,P] (f32 acc) ---------
// Bt is W pre-transposed to [P,K] f16 so both fragments load contiguously.
// One wave per 16x16 output tile; K fully unrolled in steps of 32.
template <int K, int P>
__global__ void k_gemm(const _Float16* __restrict__ A, const _Float16* __restrict__ Bt,
                       float* __restrict__ C, int N) {
  int wave = (int)((blockIdx.x * blockDim.x + threadIdx.x) >> 5);
  int lane = threadIdx.x & 31;
  constexpr int ptiles = P >> 4;
  int mt = wave / ptiles, pt = wave % ptiles;
  if (mt >= (N >> 4)) return;             // wave-uniform guard (EXEC stays all-1s)
  int half = lane >> 4, l15 = lane & 15;
  int row = (mt << 4) + l15;
  int col = (pt << 4) + l15;
  const _Float16* ar = A + (size_t)row * K;
  const _Float16* br = Bt + (size_t)col * K;
  v8f acc = {};
#pragma unroll
  for (int k = 0; k < K; k += 32) {
    // A 16x32 fragment: lane<16 holds K {0..7,16..23}, lane>=16 {8..15,24..31}
    v8h alo = *(const v8h*)(ar + k + half * 8);
    v8h ahi = *(const v8h*)(ar + k + 16 + half * 8);
    // B 32x16 fragment: lane<16 holds K {0..7,16..23}, lane>=16 {8..15,24..31}
    v8h blo = *(const v8h*)(br + k + half * 8);
    v8h bhi = *(const v8h*)(br + k + 16 + half * 8);
    v16h a, bf;
#pragma unroll
    for (int e = 0; e < 8; ++e) {
      a[e] = alo[e]; a[8 + e] = ahi[e];
      bf[e] = blo[e]; bf[8 + e] = bhi[e];
    }
    acc = __builtin_amdgcn_wmma_f32_16x16x32_f16(false, a, false, bf, (short)0, acc,
                                                 false, false);
  }
#pragma unroll
  for (int v = 0; v < 8; ++v) {
    int r = (mt << 4) + v + half * 8;
    C[(size_t)r * P + col] = acc[v];
  }
}

// ---------------- attention scores: att_i = h1 . a_self, att_j = h1 . a_nbr --
__global__ void k_att(const float* __restrict__ h1, const float* __restrict__ a_self,
                      const float* __restrict__ a_nbr, float* __restrict__ att_i,
                      float* __restrict__ att_j, int N) {
  int wave = (int)((blockIdx.x * blockDim.x + threadIdx.x) >> 5);
  int lane = threadIdx.x & 31;
  if (wave >= N) return;
  const float* hr = h1 + (size_t)wave * 128;
  float di = 0.f, dj = 0.f;
#pragma unroll
  for (int j = 0; j < 4; ++j) {
    float hv = hr[lane + j * 32];
    di += hv * a_self[lane + j * 32];
    dj += hv * a_nbr[lane + j * 32];
  }
#pragma unroll
  for (int m = 16; m >= 1; m >>= 1) { di += __shfl_xor(di, m, 32); dj += __shfl_xor(dj, m, 32); }
  if (lane == 0) { att_i[wave] = di; att_j[wave] = dj; }
}

// ---------------- GAT aggregation (one wave per node, dst-sorted CSR) -------
// Fuses: segment softmax over in-edges, alpha-weighted gather of h1[src],
// + gat_b, ReLU, BN2, emit f16 for next GEMM.
__global__ void k_gat(const float* __restrict__ h1, const float* __restrict__ att_i,
                      const float* __restrict__ att_j, const int* __restrict__ src,
                      const int* __restrict__ row_start, const float* __restrict__ gb,
                      const float* __restrict__ g2, const float* __restrict__ b2,
                      const float* __restrict__ m2, const float* __restrict__ v2,
                      _Float16* __restrict__ a2, int N) {
  int wave = (int)((blockIdx.x * blockDim.x + threadIdx.x) >> 5);
  int lane = threadIdx.x & 31;
  if (wave >= N) return;
  int s0 = row_start[wave], s1 = row_start[wave + 1];
  float ai = att_i[wave];
  // pass 1: segment max (lane-strided)
  float mx = -3.0e38f;
  for (int i = s0 + lane; i < s1; i += 32) {
    float e = ai + att_j[src[i]];
    e = e >= 0.f ? e : 0.2f * e;
    mx = fmaxf(mx, e);
  }
#pragma unroll
  for (int m = 16; m >= 1; m >>= 1) mx = fmaxf(mx, __shfl_xor(mx, m, 32));
  // pass 2: exp-sum
  float ss = 0.f;
  for (int i = s0 + lane; i < s1; i += 32) {
    float e = ai + att_j[src[i]];
    e = e >= 0.f ? e : 0.2f * e;
    ss += __expf(e - mx);
  }
#pragma unroll
  for (int m = 16; m >= 1; m >>= 1) ss += __shfl_xor(ss, m, 32);
  float inv = (s1 > s0) ? 1.f / ss : 0.f;
  // pass 3: alpha-weighted row gather; 32 lanes cover 128 channels as float4
  float4 acc; acc.x = acc.y = acc.z = acc.w = 0.f;
  for (int i = s0; i < s1; ++i) {
    int sv = src[i];
    float e = ai + att_j[sv];
    e = e >= 0.f ? e : 0.2f * e;
    float alpha = __expf(e - mx) * inv;
    float4 hv = *(const float4*)(h1 + (size_t)sv * 128 + lane * 4);
    acc.x += alpha * hv.x; acc.y += alpha * hv.y;
    acc.z += alpha * hv.z; acc.w += alpha * hv.w;
  }
  float vals[4] = {acc.x, acc.y, acc.z, acc.w};
  v4h o;
#pragma unroll
  for (int j = 0; j < 4; ++j) {
    int ch = lane * 4 + j;
    float val = vals[j] + gb[ch];
    val = val > 0.f ? val : 0.f;                                      // ReLU
    val = g2[ch] * (val - m2[ch]) * rsqrtf(v2[ch] + BN_EPS) + b2[ch]; // BN2
    o[j] = (_Float16)val;
  }
  *(v4h*)(a2 + (size_t)wave * 128 + lane * 4) = o;
}

// ---------------- GCN aggregation (one wave per node) -----------------------
// out_f16 = [BN?](ReLU(sum_e edge_w * t[src] + bias)); C in {64,32}
__global__ void k_gcn(const float* __restrict__ t, const float* __restrict__ ew,
                      const int* __restrict__ src, const int* __restrict__ row_start,
                      const float* __restrict__ bias,
                      const float* __restrict__ g, const float* __restrict__ bt,
                      const float* __restrict__ mn, const float* __restrict__ vr,
                      int use_bn, _Float16* __restrict__ outp, int N, int C) {
  int wave = (int)((blockIdx.x * blockDim.x + threadIdx.x) >> 5);
  int lane = threadIdx.x & 31;
  if (wave >= N) return;
  int s0 = row_start[wave], s1 = row_start[wave + 1];
  int cpl = C >> 5;             // channels per lane (2 or 1)
  float acc[2] = {0.f, 0.f};
  for (int i = s0; i < s1; ++i) {
    int sv = src[i];
    float w = ew[i];
    const float* tr = t + (size_t)sv * C;
    for (int j = 0; j < cpl; ++j) acc[j] += w * tr[lane + j * 32];
  }
  for (int j = 0; j < cpl; ++j) {
    int ch = lane + j * 32;
    float val = acc[j] + bias[ch];
    val = val > 0.f ? val : 0.f;
    if (use_bn) val = g[ch] * (val - mn[ch]) * rsqrtf(vr[ch] + BN_EPS) + bt[ch];
    outp[(size_t)wave * C + ch] = (_Float16)val;
  }
}

// ---------------- heads: z_mean/z_log_var/z via dual WMMA + fused epilogue --
// Wm_t / Wv_t are [64,32] f16 (transposed), so fragments load contiguously.
__global__ void k_head(const _Float16* __restrict__ H, const _Float16* __restrict__ Wm_t,
                       const _Float16* __restrict__ Wv_t, const float* __restrict__ bm,
                       const float* __restrict__ bv, const float* __restrict__ eps,
                       float* __restrict__ out, int N) {
  int wave = (int)((blockIdx.x * blockDim.x + threadIdx.x) >> 5);
  int lane = threadIdx.x & 31;
  int mt = wave >> 2, pt = wave & 3;          // P=64 -> 4 col tiles
  if (mt >= (N >> 4)) return;
  int half = lane >> 4, l15 = lane & 15;
  int row = (mt << 4) + l15;
  int col = (pt << 4) + l15;
  const _Float16* ar = H + (size_t)row * 32;  // K = 32: single WMMA step
  v8h alo = *(const v8h*)(ar + half * 8);
  v8h ahi = *(const v8h*)(ar + 16 + half * 8);
  const _Float16* pm = Wm_t + (size_t)col * 32 + half * 8;
  const _Float16* pv = Wv_t + (size_t)col * 32 + half * 8;
  v8h mlo = *(const v8h*)(pm);
  v8h mhi = *(const v8h*)(pm + 16);
  v8h vlo = *(const v8h*)(pv);
  v8h vhi = *(const v8h*)(pv + 16);
  v16h a, bmf, bvf;
#pragma unroll
  for (int e = 0; e < 8; ++e) {
    a[e] = alo[e];  a[8 + e] = ahi[e];
    bmf[e] = mlo[e]; bmf[8 + e] = mhi[e];
    bvf[e] = vlo[e]; bvf[8 + e] = vhi[e];
  }
  v8f cm = {}; v8f cv = {};
  cm = __builtin_amdgcn_wmma_f32_16x16x32_f16(false, a, false, bmf, (short)0, cm, false, false);
  cv = __builtin_amdgcn_wmma_f32_16x16x32_f16(false, a, false, bvf, (short)0, cv, false, false);
  size_t NP = (size_t)N * 64;
#pragma unroll
  for (int v = 0; v < 8; ++v) {
    int r = (mt << 4) + v + half * 8;
    size_t idx = (size_t)r * 64 + col;
    float zm = cm[v] + bm[col];
    zm = 1.f / (1.f + __expf(-zm));           // sigmoid
    float zlv = cv[v] + bv[col];
    float z = zm + __expf(0.5f * zlv) * eps[idx];
    out[idx] = zm;
    out[NP + idx] = zlv;
    out[2 * NP + idx] = z;
  }
}

// ---------------- host launcher ---------------------------------------------

extern "C" void kernel_launch(void* const* d_in, const int* in_sizes, int n_in,
                              void* d_out, int out_size, void* d_ws, size_t ws_size,
                              hipStream_t stream) {
  const float* x      = (const float*)d_in[0];
  const int*   esrc   = (const int*)d_in[1];
  const int*   edst   = (const int*)d_in[2];
  const float* ew     = (const float*)d_in[3];
  const float* bn1g   = (const float*)d_in[4];
  const float* bn1b   = (const float*)d_in[5];
  const float* bn1m   = (const float*)d_in[6];
  const float* bn1v   = (const float*)d_in[7];
  const float* gat_w  = (const float*)d_in[8];
  const float* a_self = (const float*)d_in[9];
  const float* a_nbr  = (const float*)d_in[10];
  const float* gat_b  = (const float*)d_in[11];
  const float* bn2g   = (const float*)d_in[12];
  const float* bn2b   = (const float*)d_in[13];
  const float* bn2m   = (const float*)d_in[14];
  const float* bn2v   = (const float*)d_in[15];
  const float* gcn2_w = (const float*)d_in[16];
  const float* gcn2_b = (const float*)d_in[17];
  const float* bn3g   = (const float*)d_in[18];
  const float* bn3b   = (const float*)d_in[19];
  const float* bn3m   = (const float*)d_in[20];
  const float* bn3v   = (const float*)d_in[21];
  const float* gcn3_w = (const float*)d_in[22];
  const float* gcn3_b = (const float*)d_in[23];
  const float* zm_w   = (const float*)d_in[24];
  const float* zm_b   = (const float*)d_in[25];
  const float* zv_w   = (const float*)d_in[26];
  const float* zv_b   = (const float*)d_in[27];
  const float* eps    = (const float*)d_in[28];

  const int N = in_sizes[0] / 128;
  const int E = in_sizes[1];

  // workspace layout (256B aligned regions), ~104 MB total with buffer reuse
  char* ws = (char*)d_ws;
  size_t off = 0;
  auto alloc = [&](size_t bytes) -> size_t {
    size_t r = off; off += (bytes + 255) & ~(size_t)255; return r;
  };
  _Float16* wGat = (_Float16*)(ws + alloc((size_t)128 * 128 * 2)); // [128,128] transposed
  _Float16* wG2  = (_Float16*)(ws + alloc((size_t)128 * 64 * 2));  // [64,128]
  _Float16* wG3  = (_Float16*)(ws + alloc((size_t)64 * 32 * 2));   // [32,64]
  _Float16* wZm  = (_Float16*)(ws + alloc((size_t)32 * 64 * 2));   // [64,32]
  _Float16* wZv  = (_Float16*)(ws + alloc((size_t)32 * 64 * 2));   // [64,32]
  int*   row_start = (int*)(ws + alloc((size_t)(N + 1) * 4));
  float* att_i     = (float*)(ws + alloc((size_t)N * 4));
  float* att_j     = (float*)(ws + alloc((size_t)N * 4));
  char*  bufA = ws + alloc((size_t)N * 128 * 2); // h0_f16 -> a3_f16 (N*64 f16)
  char*  bufB = ws + alloc((size_t)N * 128 * 4); // h1_f32 -> t3_f32 (N*64) -> h4_f16
  char*  bufC = ws + alloc((size_t)N * 128 * 2); // a2_f16 -> t4_f32 (N*32)
  (void)ws_size; (void)n_in; (void)out_size;

  // weights -> f16, transposed to [P,K] for contiguous B-fragment loads
  k_w_to_f16_t<<<(128 * 128 + 255) / 256, 256, 0, stream>>>(gat_w, wGat, 128, 128);
  k_w_to_f16_t<<<(128 * 64 + 255) / 256, 256, 0, stream>>>(gcn2_w, wG2, 128, 64);
  k_w_to_f16_t<<<(64 * 32 + 255) / 256, 256, 0, stream>>>(gcn3_w, wG3, 64, 32);
  k_w_to_f16_t<<<(32 * 64 + 255) / 256, 256, 0, stream>>>(zm_w, wZm, 32, 64);
  k_w_to_f16_t<<<(32 * 64 + 255) / 256, 256, 0, stream>>>(zv_w, wZv, 32, 64);

  // CSR offsets from sorted edge_dst
  k_rowstart<<<(N + 1 + 255) / 256, 256, 0, stream>>>(edst, row_start, N, E);

  // BN1 -> f16
  _Float16* h0 = (_Float16*)bufA;
  int tot = N * 128;
  k_bn1<<<(tot + 255) / 256, 256, 0, stream>>>(x, bn1g, bn1b, bn1m, bn1v, h0, tot);

  const int nodeBlocks = (N + 7) / 8;   // one wave32 per node, 8 waves/block

  // GEMM1: h1 = h0 @ gat_w   [N,128]x[128,128]
  float* h1 = (float*)bufB;
  int waves1 = (N / 16) * (128 / 16);
  k_gemm<128, 128><<<(waves1 + 7) / 8, 256, 0, stream>>>(h0, wGat, h1, N);

  // attention scores
  k_att<<<nodeBlocks, 256, 0, stream>>>(h1, a_self, a_nbr, att_i, att_j, N);

  // GAT softmax-aggregate + bias + ReLU + BN2 -> a2_f16
  _Float16* a2 = (_Float16*)bufC;
  k_gat<<<nodeBlocks, 256, 0, stream>>>(h1, att_i, att_j, esrc, row_start, gat_b,
                                        bn2g, bn2b, bn2m, bn2v, a2, N);

  // GEMM2: t3 = a2 @ gcn2_w  [N,128]x[128,64]
  float* t3 = (float*)bufB;
  int waves2 = (N / 16) * (64 / 16);
  k_gemm<128, 64><<<(waves2 + 7) / 8, 256, 0, stream>>>(a2, wG2, t3, N);

  // GCN2 aggregate + bias + ReLU + BN3 -> a3_f16
  _Float16* a3 = (_Float16*)bufA;
  k_gcn<<<nodeBlocks, 256, 0, stream>>>(t3, ew, esrc, row_start, gcn2_b,
                                        bn3g, bn3b, bn3m, bn3v, 1, a3, N, 64);

  // GEMM3: t4 = a3 @ gcn3_w  [N,64]x[64,32]
  float* t4 = (float*)bufC;
  int waves3 = (N / 16) * (32 / 16);
  k_gemm<64, 32><<<(waves3 + 7) / 8, 256, 0, stream>>>(a3, wG3, t4, N);

  // GCN3 aggregate + bias + ReLU -> h4_f16
  _Float16* h4 = (_Float16*)bufB;
  k_gcn<<<nodeBlocks, 256, 0, stream>>>(t4, ew, esrc, row_start, gcn3_b,
                                        nullptr, nullptr, nullptr, nullptr, 0, h4, N, 32);

  // heads: dual WMMA + sigmoid/exp/reparameterize -> d_out (z_mean, z_log_var, z)
  int waves4 = (N / 16) * (64 / 16);
  k_head<<<(waves4 + 7) / 8, 256, 0, stream>>>(h4, wZm, wZv, zm_b, zv_b, eps,
                                               (float*)d_out, N);
}
